// Ada_GCNResnet_45758581571745
// MI455X (gfx1250) — matrix operand order repaired and checked by reading, not compiled, not run on hardware
//
#include <hip/hip_runtime.h>

typedef __attribute__((ext_vector_type(16))) __bf16 v16bf;
typedef __attribute__((ext_vector_type(8)))  float  v8f;

__device__ __forceinline__ unsigned short f2bf(float x) {
  unsigned int u = __builtin_bit_cast(unsigned int, x);
  unsigned int r = u + 0x7FFFu + ((u >> 16) & 1u);   // round-to-nearest-even
  return (unsigned short)(r >> 16);
}

union FragBF {
  uint4 q[2];
  unsigned int u[8];
  v16bf v;
};

// ---------------------------------------------------------------------------
// NT bf16 GEMM with v_wmma_f32_16x16x32_bf16 + async global->LDS double buffer.
//   C(MxN) = A(MxK,row) * B(NxK,row)^T   [+bias] [relu]
// Outputs: fp32 C32 (MxN), bf16 C16 (MxN), bf16 transposed C16T (NxM).
// Block tile 128x128x32, 128 threads = 4 wave32 waves in a 2x2 grid;
// each wave computes a 64x64 macro-tile = 4x4 WMMA 16x16 tiles, so the
// LDS-read : WMMA ratio is 1:1 (16 ds_load_b128 per 16 XDL ops).
// Tiles staged with global_load_async_to_lds_b128 (ASYNCcnt), double-buffered;
// s_wait_asynccnt sits after the WMMAs so DMA overlaps matrix math.
// ---------------------------------------------------------------------------
template<bool HAS_BIAS, bool RELU, bool OUT32, bool OUT16, bool OUT16T>
__global__ __launch_bounds__(128) void gemm_bf16_nt(
    const unsigned short* __restrict__ A,   // M x K
    const unsigned short* __restrict__ B,   // N x K
    const float* __restrict__ bias,
    float* __restrict__ C32,
    unsigned short* __restrict__ C16,
    unsigned short* __restrict__ C16T,      // N x M
    int M, int N, int K)
{
  constexpr int BM = 128, BN = 128, BK = 32;
  __shared__ __align__(16) unsigned short lsA[2][BM * BK];  // [m][k]
  __shared__ __align__(16) unsigned short lsB[2][BN * BK];  // [n][k]

  const int tid  = threadIdx.x;
  const int lane = tid & 31;
  const int wid  = tid >> 5;
  const int wm   = wid & 1;     // 2 waves along M, 64 rows each
  const int wn   = wid >> 1;    // 2 waves along N, 64 cols each
  const int lh   = lane >> 4;   // lane half (selects K sub-block per ISA layout)
  const int l15  = lane & 15;

  const int m0 = blockIdx.y * BM;
  const int n0 = blockIdx.x * BN;

  const unsigned int* A32 = reinterpret_cast<const unsigned int*>(A);
  const unsigned int* B32 = reinterpret_cast<const unsigned int*>(B);
  const int K2 = K >> 1;        // dwords per row (K is a multiple of 32)

  // Stage one 128x32 bf16 tile (8KB) = 512 x 16B chunks, 4 per thread,
  // via per-lane async DMA straight into LDS (no VGPR round trip).
  auto stage = [&](const unsigned int* src, int r0, int rmax, int kd0,
                   unsigned short* dst) {
#pragma unroll
    for (int i = 0; i < 4; ++i) {
      int q = tid + i * 128;          // 16B chunk index
      int r = q >> 2, c = q & 3;      // row, 16B-column within row
      int rr = r0 + r; if (rr >= rmax) rr = rmax - 1;
      unsigned long long ga =
          (unsigned long long)(size_t)(src + (size_t)rr * K2 + kd0 + c * 4);
      unsigned int la = (unsigned int)(size_t)(dst + q * 8);
      asm volatile("global_load_async_to_lds_b128 %0, %1, off"
                   :: "v"(la), "v"(ga) : "memory");
    }
  };

  v8f acc[4][4];
  v8f zero = {0.f, 0.f, 0.f, 0.f, 0.f, 0.f, 0.f, 0.f};
#pragma unroll
  for (int i = 0; i < 4; ++i)
#pragma unroll
    for (int j = 0; j < 4; ++j) acc[i][j] = zero;

  // prologue: tile 0 into buffer 0
  stage(A32, m0, M, 0, lsA[0]);
  stage(B32, n0, N, 0, lsB[0]);
  asm volatile("s_wait_asynccnt 0x0" ::: "memory");
  __syncthreads();

  const int nkt = K / BK;
  for (int kt = 0; kt < nkt; ++kt) {
    const int cur = kt & 1, nxt = cur ^ 1;
    // issue next tile's async copies; they land while we do the WMMAs
    if (kt + 1 < nkt) {
      stage(A32, m0, M, (kt + 1) * 16, lsA[nxt]);
      stage(B32, n0, N, (kt + 1) * 16, lsB[nxt]);
    }

    // fragments per documented 16-bit A/B VGPR layouts; 16B ds loads
    FragBF a[4], b[4];
#pragma unroll
    for (int mt = 0; mt < 4; ++mt) {
      const uint4* p = reinterpret_cast<const uint4*>(
          lsA[cur] + (wm * 64 + mt * 16 + l15) * BK);
      a[mt].q[0] = p[lh];        // K {0-7}  / {8-15}
      a[mt].q[1] = p[2 + lh];    // K {16-23} / {24-31}
    }
#pragma unroll
    for (int nt = 0; nt < 4; ++nt) {
      const uint4* p = reinterpret_cast<const uint4*>(
          lsB[cur] + (wn * 64 + nt * 16 + l15) * BK);
      b[nt].q[0] = p[lh];        // K {0-7}  / {8-15}
      b[nt].q[1] = p[2 + lh];    // K {16-23} / {24-31}
    }
#pragma unroll
    for (int mt = 0; mt < 4; ++mt)
#pragma unroll
      for (int nt = 0; nt < 4; ++nt)
        acc[mt][nt] = __builtin_amdgcn_wmma_f32_16x16x32_bf16(
            false, a[mt].v, false, b[nt].v, (short)0, acc[mt][nt], false, false);

    asm volatile("s_wait_asynccnt 0x0" ::: "memory");
    __syncthreads();
  }

  // ---- epilogue: bias / relu / stores ----
#pragma unroll
  for (int mt = 0; mt < 4; ++mt) {
#pragma unroll
    for (int nt = 0; nt < 4; ++nt) {
      int n = n0 + wn * 64 + nt * 16 + l15;
      if (n >= N) continue;
      int mb = m0 + wm * 64 + mt * 16 + lh * 8;
      float bv = HAS_BIAS ? bias[n] : 0.0f;
      float v[8];
#pragma unroll
      for (int r = 0; r < 8; ++r) {
        float t = acc[mt][nt][r] + bv;
        v[r] = RELU ? fmaxf(t, 0.0f) : t;
      }
      if (OUT32) {
#pragma unroll
        for (int r = 0; r < 8; ++r)
          if (mb + r < M) C32[(size_t)(mb + r) * N + n] = v[r];
      }
      if (OUT16) {
#pragma unroll
        for (int r = 0; r < 8; ++r)
          if (mb + r < M) C16[(size_t)(mb + r) * N + n] = f2bf(v[r]);
      }
      if (OUT16T) {   // transposed bf16 store: 8 consecutive m -> one 16B store
        if (mb + 8 <= M) {
          uint4 q;
          unsigned int p[4];
#pragma unroll
          for (int r = 0; r < 4; ++r)
            p[r] = (unsigned)f2bf(v[2 * r]) | ((unsigned)f2bf(v[2 * r + 1]) << 16);
          q.x = p[0]; q.y = p[1]; q.z = p[2]; q.w = p[3];
          *reinterpret_cast<uint4*>(&C16T[(size_t)n * M + mb]) = q;
        } else {
#pragma unroll
          for (int r = 0; r < 8; ++r)
            if (mb + r < M) C16T[(size_t)n * M + (mb + r)] = f2bf(v[r]);
        }
      }
    }
  }
}

// ---------------------------------------------------------------------------
// Elementwise / reduction helpers
// ---------------------------------------------------------------------------
__global__ __launch_bounds__(256) void cvt_f32_bf16(
    const float* __restrict__ in, unsigned short* __restrict__ out, int n) {
  int i = blockIdx.x * 256 + threadIdx.x;
  if (i < n) out[i] = f2bf(in[i]);
}

// transpose + convert: in (Kd x Nd) fp32 -> out (Nd x Kd) bf16
__global__ __launch_bounds__(256) void cvt_transpose_bf16(
    const float* __restrict__ in, unsigned short* __restrict__ out,
    int Kd, int Nd) {
  int i = blockIdx.x * 256 + threadIdx.x;
  if (i >= Kd * Nd) return;
  int o = i % Nd, k = i / Nd;
  out[(size_t)o * Kd + k] = f2bf(in[i]);
}

// feature (B,2048,14,14) -> node-major bf16 (N=3136, C=2048)
__global__ __launch_bounds__(256) void cvt_feature_nodes(
    const float* __restrict__ f, unsigned short* __restrict__ out) {
  int i = blockIdx.x * 256 + threadIdx.x;  // over 3136*2048
  if (i >= 3136 * 2048) return;
  int n = i >> 11;
  int c = i & 2047;
  int b = n / 196, hw = n % 196;
  out[i] = f2bf(f[((size_t)b * 2048 + c) * 196 + hw]);
}

// row-normalize adj (fp32) -> bf16
__global__ __launch_bounds__(256) void rownorm_kernel(
    const float* __restrict__ adj, unsigned short* __restrict__ adjN, int N) {
  int row = blockIdx.x;
  const float* a = adj + (size_t)row * N;
  float s = 0.0f;
  for (int j = threadIdx.x; j < N; j += 256) s += a[j];
  __shared__ float red[256];
  red[threadIdx.x] = s;
  __syncthreads();
  for (int off = 128; off > 0; off >>= 1) {
    if (threadIdx.x < off) red[threadIdx.x] += red[threadIdx.x + off];
    __syncthreads();
  }
  float rinv = 1.0f / (red[0] + 1e-10f);
  unsigned short* o = adjN + (size_t)row * N;
  for (int j = threadIdx.x; j < N; j += 256) o[j] = f2bf(a[j] * rinv);
}

// fold BN (eval) + conv-up bias into scale/shift per channel
__global__ __launch_bounds__(256) void bn_params(
    const float* __restrict__ g, const float* __restrict__ bta,
    const float* __restrict__ mean, const float* __restrict__ var,
    const float* __restrict__ bup, float* __restrict__ scale,
    float* __restrict__ shift) {
  int c = blockIdx.x * 256 + threadIdx.x;
  if (c >= 2048) return;
  float s = g[c] * rsqrtf(var[c] + 1e-5f);
  scale[c] = s;
  shift[c] = bta[c] + (bup[c] - mean[c]) * s;
}

// residual + BN + 14x14 max-pool -> xpool (16,2048)
__global__ __launch_bounds__(256) void bn_res_pool(
    const float* __restrict__ feat, const float* __restrict__ xup,
    const float* __restrict__ scale, const float* __restrict__ shift,
    float* __restrict__ xpool) {
  int idx = blockIdx.x * 256 + threadIdx.x;
  if (idx >= 16 * 2048) return;
  int b = idx >> 11, c = idx & 2047;
  float sc = scale[c], sh = shift[c];
  const float* f  = feat + ((size_t)b * 2048 + c) * 196;
  const float* xu = xup + (size_t)b * 196 * 2048 + c;
  float m = -3.4e38f;
  for (int hw = 0; hw < 196; ++hw)
    m = fmaxf(m, f[hw] + xu[(size_t)hw * 2048] * sc + sh);
  xpool[idx] = m;
}

// final fc: (16,2048) x (80,2048)^T + b -> (16,80)
__global__ __launch_bounds__(256) void fc_kernel(
    const float* __restrict__ xpool, const float* __restrict__ wfc,
    const float* __restrict__ bfc, float* __restrict__ out) {
  int idx = blockIdx.x * 256 + threadIdx.x;
  if (idx >= 16 * 80) return;
  int b = idx / 80, cls = idx % 80;
  const float* x = xpool + (size_t)b * 2048;
  const float* w = wfc + (size_t)cls * 2048;
  float s = bfc[cls];
  for (int c = 0; c < 2048; ++c) s += x[c] * w[c];
  out[idx] = s;
}

// ---------------------------------------------------------------------------
extern "C" void kernel_launch(void* const* d_in, const int* in_sizes, int n_in,
                              void* d_out, int out_size, void* d_ws, size_t ws_size,
                              hipStream_t stream) {
  (void)in_sizes; (void)n_in; (void)out_size; (void)ws_size;
  const float* feature = (const float*)d_in[0];
  const float* w_down  = (const float*)d_in[1];
  const float* b_down  = (const float*)d_in[2];
  const float* w1      = (const float*)d_in[3];
  const float* b1      = (const float*)d_in[4];
  const float* w2      = (const float*)d_in[5];
  const float* b2      = (const float*)d_in[6];
  const float* w_up    = (const float*)d_in[7];
  const float* b_up    = (const float*)d_in[8];
  const float* bn_g    = (const float*)d_in[9];
  const float* bn_b    = (const float*)d_in[10];
  const float* bn_m    = (const float*)d_in[11];
  const float* bn_v    = (const float*)d_in[12];
  const float* w_fc    = (const float*)d_in[13];
  const float* b_fc    = (const float*)d_in[14];
  float* out = (float*)d_out;

  const int N = 3136, D = 1024, C = 2048;

  char* ws = (char*)d_ws;
  size_t off = 0;
  auto alloc = [&](size_t bytes) -> char* {
    char* p = ws + off;
    off += (bytes + 255) & ~(size_t)255;
    return p;
  };
  unsigned short* featB   = (unsigned short*)alloc((size_t)N * C * 2);  // node-major features
  unsigned short* feat_bf = (unsigned short*)alloc((size_t)N * D * 2);  // after conv-down
  float*          adj     = (float*)alloc((size_t)N * N * 4);
  unsigned short* adjN    = (unsigned short*)alloc((size_t)N * N * 2);
  unsigned short* gc2     = (unsigned short*)alloc((size_t)N * D * 2);
  unsigned short* wdB     = (unsigned short*)alloc((size_t)D * C * 2);
  unsigned short* w1TB    = (unsigned short*)alloc((size_t)D * D * 2);  // w1^T (out x in)
  unsigned short* w2TB    = (unsigned short*)alloc((size_t)D * D * 2);  // w2^T
  unsigned short* wuB     = (unsigned short*)alloc((size_t)C * D * 2);
  float*          scale   = (float*)alloc(C * 4);
  float*          shift   = (float*)alloc(C * 4);
  float*          xpool   = (float*)alloc(16 * C * 4);
  // aliases (lifetimes disjoint):
  unsigned short* t_bfT = featB;                  // t1^T / t2^T (D x N); featB dead after GEMM 1
  unsigned short* gc1   = featB + (size_t)N * D;  // second half of featB region
  float*          xup   = adj;                    // adj dead after rownorm

  dim3 blk(256);
  dim3 gblk(128);
  auto grd = [](int m, int n) { return dim3((n + 127) / 128, (m + 127) / 128); };

  // precision conversions
  cvt_feature_nodes<<<(N * C + 255) / 256, blk, 0, stream>>>(feature, featB);
  cvt_f32_bf16<<<(D * C + 255) / 256, blk, 0, stream>>>(w_down, wdB, D * C);
  cvt_transpose_bf16<<<(D * D + 255) / 256, blk, 0, stream>>>(w1, w1TB, D, D);
  cvt_transpose_bf16<<<(D * D + 255) / 256, blk, 0, stream>>>(w2, w2TB, D, D);
  cvt_f32_bf16<<<(C * D + 255) / 256, blk, 0, stream>>>(w_up, wuB, C * D);
  bn_params<<<(C + 255) / 256, blk, 0, stream>>>(bn_g, bn_b, bn_m, bn_v, b_up, scale, shift);

  // 1. conv-down: feat = featB * w_down^T + b_down            (bf16 out)
  gemm_bf16_nt<true, false, false, true, false>
      <<<grd(N, D), gblk, 0, stream>>>(featB, wdB, b_down, nullptr, feat_bf, nullptr, N, D, C);
  // 2. adjacency: adj = feat * feat^T                         (fp32 out)
  gemm_bf16_nt<false, false, true, false, false>
      <<<grd(N, N), gblk, 0, stream>>>(feat_bf, feat_bf, nullptr, adj, nullptr, nullptr, N, N, D);
  // 3. row-normalize -> bf16
  rownorm_kernel<<<N, blk, 0, stream>>>(adj, adjN, N);
  // 4. t1^T = (feat * w1)^T                                   (bf16 transposed out)
  gemm_bf16_nt<false, false, false, false, true>
      <<<grd(N, D), gblk, 0, stream>>>(feat_bf, w1TB, nullptr, nullptr, nullptr, t_bfT, N, D, D);
  // 5. gc1 = relu(adjN * t1 + b1)      (B = t1^T is D x N row-major)
  gemm_bf16_nt<true, true, false, true, false>
      <<<grd(N, D), gblk, 0, stream>>>(adjN, t_bfT, b1, nullptr, gc1, nullptr, N, D, N);
  // 6. t2^T = (gc1 * w2)^T
  gemm_bf16_nt<false, false, false, false, true>
      <<<grd(N, D), gblk, 0, stream>>>(gc1, w2TB, nullptr, nullptr, nullptr, t_bfT, N, D, D);
  // 7. gc2 = relu(adjN * t2 + b2)
  gemm_bf16_nt<true, true, false, true, false>
      <<<grd(N, D), gblk, 0, stream>>>(adjN, t_bfT, b2, nullptr, gc2, nullptr, N, D, N);
  // 8. conv-up: xup = gc2 * w_up^T                            (fp32 out)
  gemm_bf16_nt<false, false, true, false, false>
      <<<grd(N, C), gblk, 0, stream>>>(gc2, wuB, nullptr, xup, nullptr, nullptr, N, C, D);
  // 9. residual + BN + maxpool
  bn_res_pool<<<(16 * C + 255) / 256, blk, 0, stream>>>(feature, xup, scale, shift, xpool);
  // 10. fc
  fc_kernel<<<(16 * 80 + 255) / 256, blk, 0, stream>>>(xpool, w_fc, b_fc, out);
}
